// VarianceAdaptor_23940147708155
// MI455X (gfx1250) — matrix operand order repaired and checked by reading, not compiled, not run on hardware
//
#include <hip/hip_runtime.h>
#include <hip/hip_bf16.h>

typedef __attribute__((ext_vector_type(16))) _Float16 v16h;
typedef __attribute__((ext_vector_type(8)))  _Float16 v8h;
typedef __attribute__((ext_vector_type(4)))  _Float16 v4h;
typedef __attribute__((ext_vector_type(8)))  float    v8f;

#define BATCH   32
#define T_LEN   1024
#define HDIM    256
#define TILE_M  32
#define LDS_STR 264   // halfs per LDS row (256 + 8 pad -> 528B stride, conflict-free b128)
#define MAXM    4096
#define NROWS   (BATCH * T_LEN)

// CDNA5 async global->LDS tile fill (GLOBAL_LOAD_ASYNC_TO_LDS_B128, ASYNCcnt).
// Set to 0 to fall back to plain load+ds_store if the asm syntax needs fixing.
#define USE_ASYNC_TILE 1

// ---------------------------------------------------------------------------
// f32 -> f16 convert (4 elems/thread)
// ---------------------------------------------------------------------------
__global__ __launch_bounds__(256) void cvt_f32_f16_kernel(
    const float* __restrict__ in, _Float16* __restrict__ out, int n4)
{
    int i = blockIdx.x * 256 + threadIdx.x;
    if (i < n4) {
        float4 v = *(const float4*)(in + (size_t)i * 4);
        v4h h;
        h[0] = (_Float16)v.x; h[1] = (_Float16)v.y;
        h[2] = (_Float16)v.z; h[3] = (_Float16)v.w;
        *(v4h*)(out + (size_t)i * 4) = h;
    }
}

// ---------------------------------------------------------------------------
// conv weight [k][ci][co] f32  ->  [k][co][ci] f16  (transpose for B-fragments)
// ---------------------------------------------------------------------------
__global__ __launch_bounds__(256) void wcvt_kernel(
    const float* __restrict__ w, _Float16* __restrict__ wt)
{
    int i = blockIdx.x * 256 + threadIdx.x;       // < 3*256*256
    if (i < 3 * HDIM * HDIM) {
        int k   = i >> 16;
        int rem = i & 65535;
        int co  = rem >> 8;
        int ci  = rem & 255;
        wt[(size_t)k * 65536 + co * 256 + ci] =
            (_Float16)w[(size_t)k * 65536 + ci * 256 + co];
    }
}

// ---------------------------------------------------------------------------
// Conv1d(K=3, pad=1) as WMMA GEMM.  y = relu(conv(xh, wt) + bias), f32 out.
// Block: 8 waves = 2(M) x 4(N); tile 32 rows x 256 cols; K-loop 3 taps x 8 ci-chunks.
// ---------------------------------------------------------------------------
__global__ __launch_bounds__(256) void conv_wmma_kernel(
    const _Float16* __restrict__ xh,   // [B*T, 256] f16
    const _Float16* __restrict__ wt,   // [3][co=256][ci=256] f16
    const float*    __restrict__ bias, // [256]
    float*          __restrict__ out)  // [B*T, 256] f32
{
    __shared__ alignas(16) _Float16 tile[34 * LDS_STR];

    const int b   = blockIdx.y;
    const int t0  = blockIdx.x * TILE_M;
    const int tid = threadIdx.x;

    // Pull the weight panel toward L2 while the tile fill is in flight.
    __builtin_prefetch(wt + (size_t)tid * 768, 0, 1);

    // cooperative fill: 34 haloed rows x 256 halfs, 16B chunks, zero padding
    for (int i = tid; i < 34 * 32; i += 256) {
        int row = i >> 5;
        int ch  = i & 31;
        int t   = t0 - 1 + row;
        _Float16* lp = tile + row * LDS_STR + ch * 8;
        if (t >= 0 && t < T_LEN) {
            const _Float16* gp = xh + ((size_t)(b * T_LEN + t) * HDIM) + ch * 8;
#if USE_ASYNC_TILE
            unsigned loff = (unsigned)(uintptr_t)lp;   // low 32 bits = LDS byte address
            unsigned long long ga = (unsigned long long)(uintptr_t)gp;
            asm volatile("global_load_async_to_lds_b128 %0, %1, off"
                         :: "v"(loff), "v"(ga) : "memory");
#else
            *(v8h*)lp = *(const v8h*)gp;
#endif
        } else {
            v8h z = {};
            *(v8h*)lp = z;
        }
    }
#if USE_ASYNC_TILE
    asm volatile("s_wait_asynccnt 0x0" ::: "memory");
#endif
    __syncthreads();

    const int wave  = tid >> 5;
    const int lane  = tid & 31;
    const int waveM = wave >> 2;    // 0..1
    const int waveN = wave & 3;     // 0..3
    const int lrow  = lane & 15;
    const int khalf = lane >> 4;    // 0|1
    const int mbase = waveM * 16;

    v8f acc[4] = {};

    #pragma unroll
    for (int kt = 0; kt < 3; ++kt) {
        const _Float16* wk = wt + (size_t)kt * (HDIM * HDIM);
        #pragma unroll
        for (int cc = 0; cc < 8; ++cc) {
            // A fragment (16x32, M=time rows, K=ci chunk) from LDS.
            // lane holds row (lane&15); halfs K = kbase..+7 and kbase+16..+23.
            const _Float16* ap =
                tile + (mbase + lrow + kt) * LDS_STR + cc * 32 + khalf * 8;
            v8h alo = *(const v8h*)ap;
            v8h ahi = *(const v8h*)(ap + 16);
            v16h a;
            #pragma unroll
            for (int i = 0; i < 8; ++i) { a[i] = alo[i]; a[8 + i] = ahi[i]; }

            #pragma unroll
            for (int nt = 0; nt < 4; ++nt) {
                // B fragment (32x16): lane N = co, halfs K = khalf*8..+7 and khalf*8+16..+23
                int co = waveN * 64 + nt * 16 + lrow;
                const _Float16* bp = wk + (size_t)co * HDIM + cc * 32 + khalf * 8;
                v8h blo = *(const v8h*)bp;
                v8h bhi = *(const v8h*)(bp + 16);
                v16h bf;
                #pragma unroll
                for (int i = 0; i < 8; ++i) { bf[i] = blo[i]; bf[8 + i] = bhi[i]; }
                acc[nt] = __builtin_amdgcn_wmma_f32_16x16x32_f16(
                    false, a, false, bf, (short)0, acc[nt], false, false);
            }
        }
    }

    // epilogue: bias + relu  (C/D: lane N=co, VGPR r -> M = khalf*8 + r)
    #pragma unroll
    for (int nt = 0; nt < 4; ++nt) {
        int co = waveN * 64 + nt * 16 + lrow;
        float bv = bias[co];
        #pragma unroll
        for (int r = 0; r < 8; ++r) {
            int m = mbase + khalf * 8 + r;
            float v = acc[nt][r] + bv;
            v = v > 0.f ? v : 0.f;
            out[((size_t)(b * T_LEN + t0 + m)) * HDIM + co] = v;
        }
    }
}

// ---------------------------------------------------------------------------
// Row LayerNorm (wave per row).  LINEAR variant fuses the 256->1 projection.
// ---------------------------------------------------------------------------
template <bool LINEAR>
__global__ __launch_bounds__(256) void ln_kernel(
    const float* __restrict__ in, const float* __restrict__ g,
    const float* __restrict__ be, _Float16* __restrict__ outh,
    const float* __restrict__ lw, const float* __restrict__ lb,
    float* __restrict__ pred)
{
    const int row  = blockIdx.x * 8 + (threadIdx.x >> 5);
    const int lane = threadIdx.x & 31;
    const int c0   = lane * 8;
    const float* rp = in + (size_t)row * HDIM + c0;

    float v[8];
    *(float4*)&v[0] = *(const float4*)rp;
    *(float4*)&v[4] = *(const float4*)(rp + 4);

    float s = 0.f, sq = 0.f;
    #pragma unroll
    for (int i = 0; i < 8; ++i) { s += v[i]; sq += v[i] * v[i]; }
    #pragma unroll
    for (int off = 16; off >= 1; off >>= 1) {
        s  += __shfl_xor(s,  off, 32);
        sq += __shfl_xor(sq, off, 32);
    }
    const float mean = s * (1.f / 256.f);
    const float var  = sq * (1.f / 256.f) - mean * mean;
    const float rstd = rsqrtf(var + 1e-5f);

    if constexpr (!LINEAR) {
        v8h hv;
        #pragma unroll
        for (int i = 0; i < 8; ++i) {
            float y = (v[i] - mean) * rstd * g[c0 + i] + be[c0 + i];
            hv[i] = (_Float16)y;
        }
        *(v8h*)(outh + (size_t)row * HDIM + c0) = hv;
    } else {
        float dot = 0.f;
        #pragma unroll
        for (int i = 0; i < 8; ++i) {
            float y = (v[i] - mean) * rstd * g[c0 + i] + be[c0 + i];
            dot += y * lw[c0 + i];
        }
        #pragma unroll
        for (int off = 16; off >= 1; off >>= 1) dot += __shfl_xor(dot, off, 32);
        if (lane == 0) pred[row] = dot + lb[0];
    }
}

// ---------------------------------------------------------------------------
// bucketize(pred) -> embedding add.  Writes f32 (+ optional f16 copy).
// searchsorted side='left': first i with bins[i] >= v.
// ---------------------------------------------------------------------------
__global__ __launch_bounds__(256) void add_emb_kernel(
    const float* __restrict__ pred, const float* __restrict__ bins,
    const float* __restrict__ emb,  const float* __restrict__ xin,
    float* __restrict__ xout, _Float16* __restrict__ xh_out)
{
    const int row  = blockIdx.x * 8 + (threadIdx.x >> 5);
    const int lane = threadIdx.x & 31;
    const float pv = pred[row];

    int lo = 0, hi = 255;
    while (lo < hi) {
        int mid = (lo + hi) >> 1;
        if (bins[mid] < pv) lo = mid + 1; else hi = mid;
    }

    const int c0 = lane * 8;
    const float* xp = xin + (size_t)row * HDIM + c0;
    const float* ep = emb + (size_t)lo  * HDIM + c0;
    float4 a0 = *(const float4*)xp,      a1 = *(const float4*)(xp + 4);
    float4 e0 = *(const float4*)ep,      e1 = *(const float4*)(ep + 4);
    float o[8] = { a0.x + e0.x, a0.y + e0.y, a0.z + e0.z, a0.w + e0.w,
                   a1.x + e1.x, a1.y + e1.y, a1.z + e1.z, a1.w + e1.w };

    float* op = xout + (size_t)row * HDIM + c0;
    *(float4*)op       = make_float4(o[0], o[1], o[2], o[3]);
    *(float4*)(op + 4) = make_float4(o[4], o[5], o[6], o[7]);

    if (xh_out) {
        v8h hv;
        #pragma unroll
        for (int i = 0; i < 8; ++i) hv[i] = (_Float16)o[i];
        *(v8h*)(xh_out + (size_t)row * HDIM + c0) = hv;
    }
}

// ---------------------------------------------------------------------------
// Per-batch inclusive cumsum of durations (Hillis-Steele), mel_len as float.
// ---------------------------------------------------------------------------
__global__ void cumsum_kernel(const int* __restrict__ dur,
                              int* __restrict__ cum, float* __restrict__ mel)
{
    __shared__ int s[T_LEN];
    const int b = blockIdx.x, t = threadIdx.x;
    s[t] = dur[b * T_LEN + t];
    __syncthreads();
    for (int off = 1; off < T_LEN; off <<= 1) {
        int v = 0;
        if (t >= off) v = s[t - off];
        __syncthreads();
        if (t >= off) s[t] += v;
        __syncthreads();
    }
    cum[b * T_LEN + t] = s[t];
    if (t == T_LEN - 1) {
        int m = s[T_LEN - 1];
        mel[b] = (float)(m < MAXM ? m : MAXM);
    }
}

// ---------------------------------------------------------------------------
// Length regulation: wave per output row, searchsorted side='right' gather.
// ---------------------------------------------------------------------------
__global__ __launch_bounds__(256) void gather_kernel(
    const float* __restrict__ x, const int* __restrict__ cum,
    float* __restrict__ out)
{
    const int row  = blockIdx.x * 8 + (threadIdx.x >> 5);   // < B*MAXM
    const int lane = threadIdx.x & 31;
    const int b = row >> 12;
    const int m = row & (MAXM - 1);
    const int* c = cum + b * T_LEN;

    int lo = 0, hi = T_LEN;
    while (lo < hi) {
        int mid = (lo + hi) >> 1;
        if (c[mid] <= m) lo = mid + 1; else hi = mid;
    }
    int idx = lo < (T_LEN - 1) ? lo : (T_LEN - 1);
    int mel = c[T_LEN - 1]; mel = mel < MAXM ? mel : MAXM;
    const bool valid = m < mel;

    const int c0 = lane * 8;
    float* op = out + (size_t)row * HDIM + c0;
    if (valid) {
        const float* xp = x + ((size_t)(b * T_LEN + idx)) * HDIM + c0;
        *(float4*)op       = *(const float4*)xp;
        *(float4*)(op + 4) = *(const float4*)(xp + 4);
    } else {
        *(float4*)op       = make_float4(0.f, 0.f, 0.f, 0.f);
        *(float4*)(op + 4) = make_float4(0.f, 0.f, 0.f, 0.f);
    }
}

// ---------------------------------------------------------------------------
extern "C" void kernel_launch(void* const* d_in, const int* in_sizes, int n_in,
                              void* d_out, int out_size, void* d_ws, size_t ws_size,
                              hipStream_t stream)
{
    (void)in_sizes; (void)n_in; (void)out_size; (void)ws_size;

    const float* x     = (const float*)d_in[0];
    const int*   dur   = (const int*)d_in[1];
    const float* pemb  = (const float*)d_in[34];
    const float* eemb  = (const float*)d_in[35];
    const float* pbins = (const float*)d_in[36];
    const float* ebins = (const float*)d_in[37];

    // predictor params: [p][j], j: 0 w1,1 b1,2 g1,3 be1,4 w2,5 b2,6 g2,7 be2,8 lw,9 lb
    const float* P[3][10];
    for (int p = 0; p < 3; ++p)
        for (int j = 0; j < 10; ++j)
            P[p][j] = (const float*)d_in[4 + p * 10 + j];

    char* ws = (char*)d_ws;
    _Float16* xh    = (_Float16*)(ws);                         // 16 MB f16 x
    _Float16* h1h   = (_Float16*)(ws + ((size_t)16 << 20));    // 16 MB f16 LN1 out
    float*    convf = (float*)   (ws + ((size_t)32 << 20));    // 32 MB conv scratch
    float*    x12   = (float*)   (ws + ((size_t)64 << 20));    // 32 MB running x
    _Float16* wh    = (_Float16*)(ws + ((size_t)96 << 20));    // 6 x 384 KB f16 weights
    int*      cum   = (int*)     (ws + ((size_t)96 << 20) + (size_t)6 * 393216);

    float* outp  = (float*)d_out;                       // [32,4096,256]
    float* ldur  = outp + (size_t)BATCH * MAXM * HDIM;  // [32,1024]
    float* ppred = ldur + NROWS;
    float* epred = ppred + NROWS;
    float* mel   = epred + NROWS;                       // [32]

    // precision conversions + weight transposes
    cvt_f32_f16_kernel<<<8192, 256, 0, stream>>>(x, xh, NROWS * HDIM / 4);
    for (int p = 0; p < 3; ++p) {
        wcvt_kernel<<<768, 256, 0, stream>>>(P[p][0], wh + (size_t)(2 * p + 0) * 196608);
        wcvt_kernel<<<768, 256, 0, stream>>>(P[p][4], wh + (size_t)(2 * p + 1) * 196608);
    }
    cumsum_kernel<<<BATCH, T_LEN, 0, stream>>>(dur, cum, mel);

    auto run_pred = [&](const _Float16* xin, const float* const* pp,
                        const _Float16* w1h, const _Float16* w2h, float* pout) {
        conv_wmma_kernel<<<dim3(T_LEN / TILE_M, BATCH), 256, 0, stream>>>(xin, w1h, pp[1], convf);
        ln_kernel<false><<<NROWS / 8, 256, 0, stream>>>(convf, pp[2], pp[3], h1h,
                                                        nullptr, nullptr, nullptr);
        conv_wmma_kernel<<<dim3(T_LEN / TILE_M, BATCH), 256, 0, stream>>>(h1h, w2h, pp[5], convf);
        ln_kernel<true><<<NROWS / 8, 256, 0, stream>>>(convf, pp[6], pp[7], nullptr,
                                                       pp[8], pp[9], pout);
    };

    // duration predictor (on original x)
    run_pred(xh, P[0], wh + (size_t)0 * 196608, wh + (size_t)1 * 196608, ldur);
    // pitch predictor (on original x), then x1 = x + pitch_emb[bucket]
    run_pred(xh, P[1], wh + (size_t)2 * 196608, wh + (size_t)3 * 196608, ppred);
    add_emb_kernel<<<NROWS / 8, 256, 0, stream>>>(ppred, pbins, pemb, x, x12, xh);
    // energy predictor (on x1), then x2 = x1 + energy_emb[bucket] (in place)
    run_pred(xh, P[2], wh + (size_t)4 * 196608, wh + (size_t)5 * 196608, epred);
    add_emb_kernel<<<NROWS / 8, 256, 0, stream>>>(epred, ebins, eemb, x12, x12,
                                                  (_Float16*)nullptr);
    // length regulation
    gather_kernel<<<BATCH * MAXM / 8, 256, 0, stream>>>(x12, cum, outp);
}
